// ReinforceAttention_59949153517746
// MI455X (gfx1250) — matrix operand order repaired and checked by reading, not compiled, not run on hardware
//
#include <hip/hip_runtime.h>

#define IN_CH 64
#define HID 128
#define BATCH 4
#define HH 64
#define WW 64
#define NPIX 4096  // HH*WW

typedef __attribute__((ext_vector_type(16))) _Float16 v16h;
typedef __attribute__((ext_vector_type(8)))  _Float16 v8h;
typedef __attribute__((ext_vector_type(8)))  float    v8f;

// ---------------------------------------------------------------------------
// Kernel 1: fused QKV 1x1-conv. Writes:
//   Qh [B][N][HID] f16, pre-scaled by 1/sqrt(HID)
//   Kh [B][N][HID] f16
//   Vt [B][HID][N] f16  (transposed so PV B-matrix loads are contiguous in j)
// ---------------------------------------------------------------------------
__global__ void qkv_kernel(const float* __restrict__ x,
                           const float* __restrict__ wq, const float* __restrict__ bq,
                           const float* __restrict__ wk, const float* __restrict__ bk,
                           const float* __restrict__ wv, const float* __restrict__ bv,
                           _Float16* __restrict__ Qh, _Float16* __restrict__ Kh,
                           _Float16* __restrict__ Vt) {
    int idx = blockIdx.x * blockDim.x + threadIdx.x;   // B*HID*N threads
    int n = idx & (NPIX - 1);
    int o = (idx >> 12) & (HID - 1);
    int b = idx >> 19;

    const float* xp = x + ((size_t)b * IN_CH) * NPIX + n;
    float aq = 0.f, ak = 0.f, av = 0.f;
#pragma unroll
    for (int c = 0; c < IN_CH; ++c) {
        float xv = xp[(size_t)c * NPIX];
        aq += wq[o * IN_CH + c] * xv;
        ak += wk[o * IN_CH + c] * xv;
        av += wv[o * IN_CH + c] * xv;
    }
    const float qs = 0.08838834764831845f;  // 1/sqrt(128), folded into Q
    Qh[((size_t)b * NPIX + n) * HID + o] = (_Float16)((aq + bq[o]) * qs);
    Kh[((size_t)b * NPIX + n) * HID + o] = (_Float16)(ak + bk[o]);
    Vt[((size_t)b * HID + o) * NPIX + n] = (_Float16)(av + bv[o]);
}

// ---------------------------------------------------------------------------
// Kernel 2: flash attention. One wave32 per 16-query tile.
//   S = Q*K^T (pre-scaled), online softmax, O += P*V, O /= l.
//   Uses v_wmma_f32_16x16x32_f16; P transposed D-layout -> A-layout via LDS.
// ---------------------------------------------------------------------------
__global__ void __launch_bounds__(32)
attn_kernel(const _Float16* __restrict__ Qh, const _Float16* __restrict__ Kh,
            const _Float16* __restrict__ Vt, float* __restrict__ Obuf) {
    const int tiles_per_b = NPIX / 16;           // 256
    int b = blockIdx.x / tiles_per_b;
    int qbase = (blockIdx.x % tiles_per_b) * 16;

    int lane = threadIdx.x;
    int half = lane >> 4;                        // 0: lanes 0-15, 1: lanes 16-31
    int lr = lane & 15;

    __shared__ _Float16 Plds[16 * 32];           // 1 KB transpose staging

    // ---- load Q tile as four 16x32 f16 A-matrices (K = 0..127) ----
    v16h qa[4];
#pragma unroll
    for (int kc = 0; kc < 4; ++kc) {
        const _Float16* qp =
            Qh + ((size_t)(b * NPIX + qbase + lr) * HID + kc * 32 + 16 * half);
        v8h lo = *(const v8h*)qp;                // K = kc*32 + 16*half + 0..7
        v8h hi = *(const v8h*)(qp + 8);          // K = kc*32 + 16*half + 8..15
#pragma unroll
        for (int i = 0; i < 8; ++i) { qa[kc][i] = lo[i]; qa[kc][8 + i] = hi[i]; }
    }

    v8f O[8];                                    // 16 rows x 128 cols f32 accum
#pragma unroll
    for (int c = 0; c < 8; ++c) O[c] = {};
    float m_old[8], l[8];
#pragma unroll
    for (int r = 0; r < 8; ++r) { m_old[r] = -1e30f; l[r] = 0.f; }

    for (int jb = 0; jb < NPIX; jb += 32) {      // 128 key-block iterations
        // ---- S tiles: keys jb..jb+15 (s0) and jb+16..jb+31 (s1) ----
        v8f s0 = {}, s1 = {};
#pragma unroll
        for (int kc = 0; kc < 4; ++kc) {
            v16h bk0 = *(const v16h*)(Kh + ((size_t)(b * NPIX + jb + lr) * HID
                                            + kc * 32 + 16 * half));
            s0 = __builtin_amdgcn_wmma_f32_16x16x32_f16(false, qa[kc], false, bk0,
                                                        (short)0, s0, false, false);
            v16h bk1 = *(const v16h*)(Kh + ((size_t)(b * NPIX + jb + 16 + lr) * HID
                                            + kc * 32 + 16 * half));
            s1 = __builtin_amdgcn_wmma_f32_16x16x32_f16(false, qa[kc], false, bk1,
                                                        (short)0, s1, false, false);
        }

        // ---- online softmax: row stats via 16-lane xor-shuffle reductions ----
        float corr[8];
#pragma unroll
        for (int r = 0; r < 8; ++r) {
            float mx = fmaxf(s0[r], s1[r]);
            mx = fmaxf(mx, __shfl_xor(mx, 1));
            mx = fmaxf(mx, __shfl_xor(mx, 2));
            mx = fmaxf(mx, __shfl_xor(mx, 4));
            mx = fmaxf(mx, __shfl_xor(mx, 8));
            float m_new = fmaxf(m_old[r], mx);
            corr[r] = __expf(m_old[r] - m_new);
            float p0 = __expf(s0[r] - m_new);
            float p1 = __expf(s1[r] - m_new);
            s0[r] = p0; s1[r] = p1;
            float rs = p0 + p1;
            rs += __shfl_xor(rs, 1);
            rs += __shfl_xor(rs, 2);
            rs += __shfl_xor(rs, 4);
            rs += __shfl_xor(rs, 8);
            l[r] = l[r] * corr[r] + rs;
            m_old[r] = m_new;
        }
        // rescale running accumulator row-wise
#pragma unroll
        for (int c = 0; c < 8; ++c)
#pragma unroll
            for (int r = 0; r < 8; ++r) O[c][r] *= corr[r];

        // ---- transpose P (f32 D-layout -> f16 16x32 A-layout) through LDS ----
#pragma unroll
        for (int r = 0; r < 8; ++r) {
            int m = r + 8 * half;
            Plds[m * 32 + lr]      = (_Float16)s0[r];
            Plds[m * 32 + 16 + lr] = (_Float16)s1[r];
        }
        __syncthreads();
        v8h plo = *(const v8h*)(Plds + lr * 32 + 16 * half);
        v8h phi = *(const v8h*)(Plds + lr * 32 + 16 * half + 8);
        v16h pA;
#pragma unroll
        for (int i = 0; i < 8; ++i) { pA[i] = plo[i]; pA[8 + i] = phi[i]; }
        __syncthreads();

        // ---- O += P * V  (8 WMMAs, one per 16-wide d chunk) ----
#pragma unroll
        for (int c = 0; c < 8; ++c) {
            v16h bv = *(const v16h*)(Vt + ((size_t)(b * HID + 16 * c + lr) * NPIX
                                           + jb + 16 * half));
            O[c] = __builtin_amdgcn_wmma_f32_16x16x32_f16(false, pA, false, bv,
                                                          (short)0, O[c], false, false);
        }
    }

    // ---- finalize: O /= l, store [B][N][HID] f32 ----
    float inv_l[8];
#pragma unroll
    for (int r = 0; r < 8; ++r) inv_l[r] = 1.0f / l[r];
#pragma unroll
    for (int c = 0; c < 8; ++c)
#pragma unroll
        for (int r = 0; r < 8; ++r) {
            int m = r + 8 * half;
            int d = 16 * c + lr;
            Obuf[((size_t)(b * NPIX + qbase + m)) * HID + d] = O[c][r] * inv_l[r];
        }
}

// ---------------------------------------------------------------------------
// Kernel 3: output projection + residual + CBAM channel mean/max.
//   One block of 64 threads per (b, n) pixel; thread c owns output channel c.
// ---------------------------------------------------------------------------
__global__ void proj_kernel(const float* __restrict__ Obuf, const float* __restrict__ x,
                            const float* __restrict__ wo, const float* __restrict__ bo,
                            const float* __restrict__ alpha_p,
                            float* __restrict__ outbuf,
                            float* __restrict__ meanbuf, float* __restrict__ maxbuf) {
    int bn = blockIdx.x;                 // b*N + n
    int b = bn >> 12;
    int n = bn & (NPIX - 1);
    int c = threadIdx.x;                 // 0..63

    __shared__ float orow[HID];
    __shared__ float sred[IN_CH];
    __shared__ float smax[IN_CH];

    orow[c]      = Obuf[(size_t)bn * HID + c];
    orow[c + 64] = Obuf[(size_t)bn * HID + 64 + c];
    __syncthreads();

    float acc = bo[c];
#pragma unroll 16
    for (int d = 0; d < HID; ++d) acc += wo[c * HID + d] * orow[d];

    float alpha = *alpha_p;
    float val = alpha * acc + (1.0f - alpha) * x[((size_t)b * IN_CH + c) * NPIX + n];
    outbuf[((size_t)b * IN_CH + c) * NPIX + n] = val;

    sred[c] = val; smax[c] = val;
    __syncthreads();
#pragma unroll
    for (int s = 32; s > 0; s >>= 1) {
        if (c < s) {
            sred[c] += sred[c + s];
            smax[c] = fmaxf(smax[c], smax[c + s]);
        }
        __syncthreads();
    }
    if (c == 0) { meanbuf[bn] = sred[0] * (1.0f / IN_CH); maxbuf[bn] = smax[0]; }
}

// ---------------------------------------------------------------------------
// Kernel 4: 7x7 conv over [mean,max] + sigmoid -> gate[b][n]
// ---------------------------------------------------------------------------
__global__ void gate_kernel(const float* __restrict__ meanbuf,
                            const float* __restrict__ maxbuf,
                            const float* __restrict__ ws, const float* __restrict__ bs,
                            float* __restrict__ gate) {
    int idx = blockIdx.x * blockDim.x + threadIdx.x;   // B*N
    if (idx >= BATCH * NPIX) return;
    int b = idx >> 12;
    int hw = idx & (NPIX - 1);
    int h = hw >> 6, w = hw & 63;

    float acc = bs[0];
#pragma unroll
    for (int dy = -3; dy <= 3; ++dy) {
        int yy = h + dy;
        if ((unsigned)yy >= (unsigned)HH) continue;
#pragma unroll
        for (int dx = -3; dx <= 3; ++dx) {
            int xx = w + dx;
            if ((unsigned)xx >= (unsigned)WW) continue;
            int p = (yy << 6) + xx;
            int wi = (dy + 3) * 7 + (dx + 3);
            acc += ws[wi]      * meanbuf[(b << 12) + p]
                 + ws[49 + wi] * maxbuf[(b << 12) + p];
        }
    }
    gate[idx] = 1.0f / (1.0f + __expf(-acc));
}

// ---------------------------------------------------------------------------
// Kernel 5: elementwise gating -> d_out
// ---------------------------------------------------------------------------
__global__ void final_kernel(const float* __restrict__ outbuf,
                             const float* __restrict__ gate,
                             float* __restrict__ out) {
    int idx = blockIdx.x * blockDim.x + threadIdx.x;   // B*C*N
    if (idx >= BATCH * IN_CH * NPIX) return;
    int n = idx & (NPIX - 1);
    int b = idx >> 18;                                  // C*N = 2^18
    out[idx] = outbuf[idx] * gate[(b << 12) + n];
}

// ---------------------------------------------------------------------------
extern "C" void kernel_launch(void* const* d_in, const int* in_sizes, int n_in,
                              void* d_out, int out_size, void* d_ws, size_t ws_size,
                              hipStream_t stream) {
    const float* x     = (const float*)d_in[0];
    const float* wq    = (const float*)d_in[1];
    const float* bq    = (const float*)d_in[2];
    const float* wk    = (const float*)d_in[3];
    const float* bk    = (const float*)d_in[4];
    const float* wv    = (const float*)d_in[5];
    const float* bv    = (const float*)d_in[6];
    const float* wo    = (const float*)d_in[7];
    const float* bo    = (const float*)d_in[8];
    const float* wsp   = (const float*)d_in[9];
    const float* bs    = (const float*)d_in[10];
    const float* alpha = (const float*)d_in[11];

    char* wsb = (char*)d_ws;
    _Float16* Qh   = (_Float16*)(wsb);                                   // 4 MB
    _Float16* Kh   = (_Float16*)(wsb + ((size_t)4  << 20));              // 4 MB
    _Float16* Vt   = (_Float16*)(wsb + ((size_t)8  << 20));              // 4 MB
    float*    Obuf = (float*)   (wsb + ((size_t)12 << 20));              // 8 MB
    float*    outb = (float*)   (wsb + ((size_t)20 << 20));              // 4 MB
    float*    mnb  = (float*)   (wsb + ((size_t)24 << 20));              // 64 KB
    float*    mxb  = (float*)   (wsb + ((size_t)24 << 20) + (64u << 10));
    float*    gate = (float*)   (wsb + ((size_t)24 << 20) + (128u << 10));

    qkv_kernel<<<(BATCH * HID * NPIX) / 256, 256, 0, stream>>>(
        x, wq, bq, wk, bk, wv, bv, Qh, Kh, Vt);

    attn_kernel<<<BATCH * (NPIX / 16), 32, 0, stream>>>(Qh, Kh, Vt, Obuf);

    proj_kernel<<<BATCH * NPIX, IN_CH, 0, stream>>>(
        Obuf, x, wo, bo, alpha, outb, mnb, mxb);

    gate_kernel<<<(BATCH * NPIX + 255) / 256, 256, 0, stream>>>(mnb, mxb, wsp, bs, gate);

    final_kernel<<<(BATCH * IN_CH * NPIX + 255) / 256, 256, 0, stream>>>(
        outb, gate, (float*)d_out);
}
